// graphsage_60533269070026
// MI455X (gfx1250) — matrix operand order repaired and checked by reading, not compiled, not run on hardware
//
#include <hip/hip_runtime.h>

#define NN 16384
#define AS 68      // LDS A stride in f32  (64 + 4 pad  -> 272B rows, 16B aligned)
#define BS 72      // LDS B stride in bf16 (64 + 8 pad  -> 144B rows, 16B aligned)
#define A2S 136    // LDS concat-A stride in bf16 (128 + 8 pad -> 272B rows)
#define WS2 136    // LDS W^T stride in bf16

typedef __bf16 bf16_t;
typedef bf16_t v16bf __attribute__((ext_vector_type(16)));
typedef float v8f __attribute__((ext_vector_type(8)));
typedef float f32x4 __attribute__((ext_vector_type(4)));
typedef unsigned short u16x8 __attribute__((ext_vector_type(8)));
typedef unsigned short u16x16 __attribute__((ext_vector_type(16)));
typedef unsigned int u32x4 __attribute__((ext_vector_type(4)));
typedef int i32x4 __attribute__((ext_vector_type(4)));
typedef int i32x8 __attribute__((ext_vector_type(8)));

union BFrag { u16x16 u; v16bf bf; };

#if defined(__has_builtin)
#if __has_builtin(__builtin_amdgcn_tensor_load_to_lds)
#define USE_TDM 1
#endif
#endif
#ifndef USE_TDM
#define USE_TDM 0
#endif

// Native f32 -> bf16 conversion; let ISel pick v_cvt_pk_bf16_f32 / scalar cvt.
__device__ __forceinline__ bf16_t cvt_bf(float f) { return (bf16_t)f; }

__device__ __forceinline__ unsigned short f2bf(float f) {
  union { bf16_t b; unsigned short u; } t;
  t.b = (bf16_t)f;
  return t.u;
}

// A-matrix fragment (16x32 bf16): lane L = row L%16; elems 0..7 -> K = 8*(L/16)+0..7,
// elems 8..15 -> K = 16 + 8*(L/16)+0..7  (ISA 7.12.2, 16-bit A 16x32).
__device__ __forceinline__ v16bf afrag_from_f32(const float* base, int stride, int lane, int kb) {
  int r = lane & 15;
  int kh = (lane >> 4) << 3;
  const float* p = base + r * stride + kb + kh;
  f32x4 f0 = *(const f32x4*)p;
  f32x4 f1 = *(const f32x4*)(p + 4);
  f32x4 f2 = *(const f32x4*)(p + 16);
  f32x4 f3 = *(const f32x4*)(p + 20);
  v16bf fr;
#pragma unroll
  for (int i = 0; i < 4; ++i) {
    fr[i]      = cvt_bf(f0[i]);
    fr[4 + i]  = cvt_bf(f1[i]);
    fr[8 + i]  = cvt_bf(f2[i]);
    fr[12 + i] = cvt_bf(f3[i]);
  }
  return fr;
}

__device__ __forceinline__ v16bf afrag_from_bf16(const unsigned short* base, int stride, int lane, int kb) {
  int r = lane & 15;
  int kh = (lane >> 4) << 3;
  const unsigned short* p = base + r * stride + kb + kh;
  u16x8 lo = *(const u16x8*)p;
  u16x8 hi = *(const u16x8*)(p + 16);
  BFrag fr;
#pragma unroll
  for (int i = 0; i < 8; ++i) { fr.u[i] = lo[i]; fr.u[8 + i] = hi[i]; }
  return fr.bf;
}

// B-matrix fragment (32x16 bf16): lane L = col L%16; elems 0..7 -> K = 8*(L/16)+0..7, elems 8..15 -> K = 16 + 8*(L/16)+0..7.
__device__ __forceinline__ v16bf bfrag_from_bf16(const unsigned short* base, int stride, int lane, int kb) {
  int n = lane & 15;
  int kh = (lane >> 4) << 3;
  const unsigned short* p = base + n * stride + kb + kh;
  u16x8 lo = *(const u16x8*)p;
  u16x8 hi = *(const u16x8*)(p + 16);
  BFrag fr;
#pragma unroll
  for (int i = 0; i < 8; ++i) { fr.u[i] = lo[i]; fr.u[8 + i] = hi[i]; }
  return fr.bf;
}

__device__ __forceinline__ v8f wmma_bf16(v16bf a, v16bf b, v8f c) {
  return __builtin_amdgcn_wmma_f32_16x16x32_bf16(false, a, false, b, (short)0, c, false, false);
}

#if USE_TDM
// 2D TDM descriptor per CDNA5 ISA ch.8 (D# groups 0/1; groups 2/3 zero for 2D tiles).
// 6-arg builtin form (clang-23 / therock-10.0): (u32x4, i32x8, i32x4, i32x4, i32x8, i32 cpol).
__device__ __forceinline__ void tdm_load_2d(unsigned lds, const void* gptr, unsigned dsz,
                                            unsigned t0, unsigned t1, unsigned td0, unsigned td1,
                                            unsigned long long s0, unsigned pint, unsigned pamt,
                                            unsigned pen) {
  unsigned long long ga = (unsigned long long)gptr;
  u32x4 g0;
  g0[0] = 1u;  // count = 1 valid descriptor
  g0[1] = (unsigned)__builtin_amdgcn_readfirstlane((int)lds);
  g0[2] = (unsigned)__builtin_amdgcn_readfirstlane((int)(unsigned)ga);
  g0[3] = ((unsigned)__builtin_amdgcn_readfirstlane((int)((unsigned)(ga >> 32) & 0x01FFFFFFu))) |
          (2u << 30);  // type = 2 ("image")
  i32x8 g1;
  g1[0] = (int)((dsz << 16) | (pen << 20) | (pint << 22) | (pamt << 25));
  g1[1] = (int)((td0 & 0xFFFFu) << 16);
  g1[2] = (int)((td0 >> 16) | ((td1 & 0xFFFFu) << 16));
  g1[3] = (int)((td1 >> 16) | (t0 << 16));
  g1[4] = (int)(t1 & 0xFFFFu);
  g1[5] = (int)(s0 & 0xFFFFFFFFull);
  g1[6] = (int)((s0 >> 32) & 0xFFFFull);
  g1[7] = 0;
  i32x4 z4 = {0, 0, 0, 0};
  i32x8 z8 = {0, 0, 0, 0, 0, 0, 0, 0};
  __builtin_amdgcn_tensor_load_to_lds(g0, g1, z4, z4, z8, 0);
}
#endif

// ---------------------------------------------------------------------------
// x [NN,64] f32 -> xT [64,NN] bf16 (tiled transpose through LDS)
__global__ __launch_bounds__(256) void xpose_bf16_kernel(const float* __restrict__ x,
                                                         unsigned short* __restrict__ xT) {
  __shared__ unsigned short tile[64][72];
  int k0 = blockIdx.x * 64;
  int t = threadIdx.x;
  int r = t >> 2;
  int c0 = (t & 3) * 16;
  const float* src = x + (size_t)(k0 + r) * 64 + c0;
#pragma unroll
  for (int i = 0; i < 16; ++i) tile[r][c0 + i] = f2bf(src[i]);
  __syncthreads();
  unsigned short* dst = xT + (size_t)r * NN + k0 + c0;
#pragma unroll
  for (int i = 0; i < 16; ++i) dst[i] = tile[c0 + i][r];
}

// ---------------------------------------------------------------------------
// One GraphSAGE layer: support = adj @ xin ; hout = relu(concat([xin,support]) @ W)
// Block = 128 threads (4 waves); tile = 64 rows x 64 cols; K-chunk = 64.
__global__ __launch_bounds__(128) void sage_layer_kernel(
    const float* __restrict__ adj,          // [NN,NN] f32
    const unsigned short* __restrict__ xT,  // [64,NN] bf16 (transposed input)
    const float* __restrict__ xrow,         // [NN,64] f32 (row-major input, for concat)
    const float* __restrict__ W,            // [128,64] f32
    unsigned short* __restrict__ houtT,     // [64,NN] bf16
    float* __restrict__ hout)               // [NN,64] f32
{
  __shared__ float ldsA[2][64 * AS];           // adj tiles (f32), double-buffered
  __shared__ unsigned short ldsB[2][64 * BS];  // xT tiles (bf16), double-buffered
  __shared__ unsigned short ldsW[64 * WS2];    // W^T (bf16), resident
  unsigned short* ldsA2 = (unsigned short*)&ldsA[0][0];  // phase-2 concat tile (aliased)

  const int tid = threadIdx.x;
  const int w = tid >> 5;       // wave 0..3 -> rows 16w..16w+15 of the block tile
  const int lane = tid & 31;
  const int row0 = blockIdx.x * 64;

  // Stage W^T into LDS once (bf16). Covered by the first barrier below.
  for (int i = tid; i < 128 * 64; i += 128) {
    int k = i >> 6, n = i & 63;
    ldsW[n * WS2 + k] = f2bf(W[i]);
  }

  v8f acc[4] = {};

  auto compute_chunk = [&](int buf) {
    const float* abase = &ldsA[buf][(w << 4) * AS];
#pragma unroll
    for (int kb = 0; kb < 64; kb += 32) {
      v16bf a = afrag_from_f32(abase, AS, lane, kb);
#pragma unroll
      for (int nt = 0; nt < 4; ++nt) {
        v16bf b = bfrag_from_bf16(&ldsB[buf][(nt << 4) * BS], BS, lane, kb);
        acc[nt] = wmma_bf16(a, b, acc[nt]);
      }
    }
  };

#if USE_TDM
  auto issue = [&](int kc, int buf) {
    // A: this wave's 16 adj rows x 64 K-cols (f32, data_size=2 -> 4B),
    //    pad 4 DWORDs every 64 DWORDs -> LDS row stride 68 f32.
    const float* ga = adj + (size_t)(row0 + (w << 4)) * NN + kc;
    unsigned la = (unsigned)(unsigned long long)&ldsA[buf][(w << 4) * AS];
    tdm_load_2d(la, ga, 2u, 64u, 16u, NN, NN, (unsigned long long)NN, 5u, 3u, 1u);
    if (w == 0) {
      // B: 64 feature-rows of xT x 64 K-cols (bf16, data_size=1 -> 2B),
      //    pad 4 DWORDs every 32 DWORDs -> LDS row stride 72 bf16.
      const unsigned short* gb = xT + kc;
      unsigned lb = (unsigned)(unsigned long long)&ldsB[buf][0];
      tdm_load_2d(lb, gb, 1u, 64u, 64u, NN, 64u, (unsigned long long)NN, 4u, 3u, 1u);
    }
  };

  issue(0, 0);
  __builtin_amdgcn_s_wait_tensorcnt(0);
  __syncthreads();
  for (int kc = 0; kc < NN; kc += 64) {
    int cur = (kc >> 6) & 1;
    if (kc + 64 < NN) issue(kc + 64, cur ^ 1);   // TDM overlaps with WMMA below
    compute_chunk(cur);
    if (kc + 64 < NN) __builtin_amdgcn_s_wait_tensorcnt(0);
    __syncthreads();
  }
#else
  auto stage = [&](int kc, int buf) {
    int r = tid >> 1, c0 = (tid & 1) * 32;
    const float* src = adj + (size_t)(row0 + r) * NN + kc + c0;
    float* dst = &ldsA[buf][r * AS + c0];
#pragma unroll
    for (int i = 0; i < 8; ++i) *(f32x4*)(dst + 4 * i) = *(const f32x4*)(src + 4 * i);
    const unsigned short* sb = xT + (size_t)r * NN + kc + c0;
    unsigned short* db = &ldsB[buf][r * BS + c0];
#pragma unroll
    for (int i = 0; i < 4; ++i) *(u16x8*)(db + 8 * i) = *(const u16x8*)(sb + 8 * i);
  };
  for (int kc = 0; kc < NN; kc += 64) {
    __syncthreads();
    stage(kc, 0);
    __syncthreads();
    compute_chunk(0);
  }
  __syncthreads();
#endif

  // ---- Phase 2: hout = relu(concat([xin, support]) @ W), K2 = 128, all WMMA ----
  // Build concat tile in LDS (bf16): features 0..63 = xin rows, 64..127 = support.
  {
    int r = tid >> 1, c0 = (tid & 1) * 32;
    const float* src = xrow + (size_t)(row0 + r) * 64 + c0;
    unsigned short* dst = ldsA2 + r * A2S + c0;
#pragma unroll
    for (int i = 0; i < 32; ++i) dst[i] = f2bf(src[i]);
  }
#pragma unroll
  for (int nt = 0; nt < 4; ++nt)
#pragma unroll
    for (int j = 0; j < 8; ++j) {
      int r = (w << 4) + ((lane >> 4) << 3) + j;   // C layout: VGPR j -> M = j + 8*(lane/16)
      int c = (nt << 4) + (lane & 15);
      ldsA2[r * A2S + 64 + c] = f2bf(acc[nt][j]);
    }
  __syncthreads();

  v8f acc2[4] = {};
  const unsigned short* a2base = ldsA2 + (w << 4) * A2S;
#pragma unroll
  for (int kb = 0; kb < 128; kb += 32) {
    v16bf a = afrag_from_bf16(a2base, A2S, lane, kb);
#pragma unroll
    for (int nt = 0; nt < 4; ++nt) {
      v16bf b = bfrag_from_bf16(ldsW + (nt << 4) * WS2, WS2, lane, kb);
      acc2[nt] = wmma_bf16(a, b, acc2[nt]);
    }
  }

#pragma unroll
  for (int nt = 0; nt < 4; ++nt)
#pragma unroll
    for (int j = 0; j < 8; ++j) {
      float v = acc2[nt][j];
      v = v > 0.0f ? v : 0.0f;   // ReLU
      int r = (w << 4) + ((lane >> 4) << 3) + j;
      int c = (nt << 4) + (lane & 15);
      size_t rg = (size_t)(row0 + r);
      hout[rg * 64 + c] = v;
      houtT[(size_t)c * NN + rg] = f2bf(v);
    }
}

// ---------------------------------------------------------------------------
// logits = h @ W_lin^T ; out = log_softmax(logits)   (tiny: one thread per node)
__global__ __launch_bounds__(256) void classifier_kernel(const float* __restrict__ h,
                                                         const float* __restrict__ Wlin,
                                                         float* __restrict__ out) {
  __shared__ float wl[16 * 64];
  int tid = threadIdx.x;
  for (int i = tid; i < 16 * 64; i += 256) wl[i] = Wlin[i];
  __syncthreads();
  int row = blockIdx.x * 256 + tid;
  const float* hr = h + (size_t)row * 64;
  float hb[64];
#pragma unroll
  for (int i = 0; i < 64; ++i) hb[i] = hr[i];
  float logits[16];
#pragma unroll
  for (int c = 0; c < 16; ++c) {
    float s = 0.0f;
#pragma unroll
    for (int k = 0; k < 64; ++k) s = fmaf(hb[k], wl[c * 64 + k], s);
    logits[c] = s;
  }
  float m = logits[0];
#pragma unroll
  for (int c = 1; c < 16; ++c) m = fmaxf(m, logits[c]);
  float s = 0.0f;
#pragma unroll
  for (int c = 0; c < 16; ++c) s += __expf(logits[c] - m);
  float ls = __logf(s);
  float* o = out + (size_t)row * 16;
#pragma unroll
  for (int c = 0; c < 16; ++c) o[c] = logits[c] - m - ls;
}

// ---------------------------------------------------------------------------
extern "C" void kernel_launch(void* const* d_in, const int* in_sizes, int n_in,
                              void* d_out, int out_size, void* d_ws, size_t ws_size,
                              hipStream_t stream) {
  const float* x    = (const float*)d_in[0];  // [16384,64]
  const float* adj  = (const float*)d_in[1];  // [16384,16384]
  const float* W1   = (const float*)d_in[2];  // [128,64]
  const float* W2   = (const float*)d_in[3];  // [128,64]
  const float* Wlin = (const float*)d_in[4];  // [16,64]
  float* out = (float*)d_out;                 // [16384,16]

  char* ws = (char*)d_ws;
  unsigned short* xT  = (unsigned short*)(ws);                        // 2 MB bf16 [64,NN]
  unsigned short* h1T = (unsigned short*)(ws + (size_t)(2u << 20));   // 2 MB
  unsigned short* h2T = (unsigned short*)(ws + (size_t)(4u << 20));   // 2 MB (unused sink)
  float* h1 = (float*)(ws + (size_t)(6u << 20));                      // 4 MB f32 [NN,64]
  float* h2 = (float*)(ws + (size_t)(10u << 20));                     // 4 MB

  xpose_bf16_kernel<<<NN / 64, 256, 0, stream>>>(x, xT);
  sage_layer_kernel<<<NN / 64, 128, 0, stream>>>(adj, xT, x, W1, h1T, h1);
  sage_layer_kernel<<<NN / 64, 128, 0, stream>>>(adj, h1T, h1, W2, h2T, h2);
  classifier_kernel<<<NN / 256, 256, 0, stream>>>(h2, Wlin, out);
}
